// Model_2637109920294
// MI455X (gfx1250) — matrix operand order repaired and checked by reading, not compiled, not run on hardware
//
#include <hip/hip_runtime.h>
#include <hip/hip_bf16.h>

typedef __attribute__((ext_vector_type(16))) _Float16 v16h;
typedef __attribute__((ext_vector_type(8)))  float    v8f;

static __device__ __forceinline__ int imin(int a, int b) { return a < b ? a : b; }
static __device__ __forceinline__ int imax(int a, int b) { return a > b ? a : b; }
static __device__ __forceinline__ float lrelu_f(float x) { return x >= 0.f ? x : 0.05f * x; }

// ---------------------------------------------------------------------------
// BN folding: inv = g / sqrt(v + eps), beta = b - m * inv
// ---------------------------------------------------------------------------
__global__ void bn_prep_kernel(const float* g, const float* b, const float* m,
                               const float* v, float* inv, float* beta, int C) {
  int i = blockIdx.x * blockDim.x + threadIdx.x;
  if (i < C) {
    float iv = g[i] * rsqrtf(v[i] + 1e-5f);
    inv[i] = iv;
    beta[i] = b[i] - m[i] * iv;
  }
}

// ---------------------------------------------------------------------------
// Weight transforms.  conv3x3 + avgpool3(stride3) == conv5x5 stride3 with
// weff[u,v] = (1/9) * sum of w3 over kh in [u-2,u]&[0,2], kw likewise.
// ---------------------------------------------------------------------------
// Layer0 (Cin=6): padded row-segment layout  k = u*32 + (v*6 + ci), K=160.
__global__ void wt_sphere0_kernel(const float* w3, _Float16* w5) {
  int t = blockIdx.x * blockDim.x + threadIdx.x;
  if (t >= 32 * 5 * 32) return;
  int slot = t & 31; int t2 = t >> 5;
  int uu = t2 % 5;   int co = t2 / 5;
  float s = 0.f;
  if (slot < 30) {
    int vv = slot / 6, ci = slot % 6;
    for (int kh = imax(0, uu - 2); kh <= imin(2, uu); ++kh)
      for (int kw = imax(0, vv - 2); kw <= imin(2, vv); ++kw)
        s += w3[((co * 6 + ci) * 3 + kh) * 3 + kw];
  }
  w5[(size_t)co * 160 + uu * 32 + slot] = (_Float16)(s * (1.f / 9.f));
}

// Layer1 (Cin=32): standard layout  k = (u*5+v)*Cin + ci.
__global__ void wt_sphere_kernel(const float* w3, _Float16* w5, int Cout, int Cin) {
  int t = blockIdx.x * blockDim.x + threadIdx.x;
  int total = Cout * 25 * Cin;
  if (t >= total) return;
  int ci = t % Cin; int t2 = t / Cin;
  int vv = t2 % 5;  int t3 = t2 / 5;
  int uu = t3 % 5;  int co = t3 / 5;
  float s = 0.f;
  for (int kh = imax(0, uu - 2); kh <= imin(2, uu); ++kh)
    for (int kw = imax(0, vv - 2); kw <= imin(2, vv); ++kw)
      s += w3[((co * Cin + ci) * 3 + kh) * 3 + kw];
  w5[((size_t)co * 25 + uu * 5 + vv) * Cin + ci] = (_Float16)(s * (1.f / 9.f));
}

// OIHW f32 -> [co][(kh*KW+kw)*Cin+ci] f16
__global__ void wt_conv_kernel(const float* w, _Float16* wo, int Cout, int Cin,
                               int KH, int KW) {
  int t = blockIdx.x * blockDim.x + threadIdx.x;
  int total = Cout * Cin * KH * KW;
  if (t >= total) return;
  int ci = t % Cin; int t2 = t / Cin;
  int kw = t2 % KW; int t3 = t2 / KW;
  int kh = t3 % KH; int co = t3 / KH;
  wo[t] = (_Float16)w[((co * Cin + ci) * KH + kh) * KW + kw];
}

// proj_w is [256][c*20 + y*10 + x] (NCHW flatten); conv5 out is NHWC flatten.
__global__ void wt_proj_kernel(const float* pw, _Float16* pt) {
  int t = blockIdx.x * blockDim.x + threadIdx.x;
  if (t >= 256 * 2560) return;
  int k = t % 2560; int n = t / 2560;
  int c = k & 127;  int yx = k >> 7;
  pt[t] = (_Float16)pw[n * 2560 + c * 20 + yx];
}

__global__ void f32_to_f16_kernel(const float* in, _Float16* out, int n) {
  int t = blockIdx.x * blockDim.x + threadIdx.x;
  if (t < n) out[t] = (_Float16)in[t];
}

// ---------------------------------------------------------------------------
// Spherical resampling gather. Grid coords are exact integers -> bilinear
// degenerates to a gather; compute (lat,lon) analytically.
// ---------------------------------------------------------------------------
static __device__ __forceinline__ void sphere_idx(int i, int j, int H, int W,
                                                  int Wout3, int* lat, int* lon) {
  int flat = i * Wout3 + j;
  int Wout9 = Wout3 * 3;
  int ho = flat / Wout9;
  int rem = flat - ho * Wout9;
  int wo = rem / 9;
  int r2 = rem - wo * 9;
  int kh = r2 / 3, kw = r2 - kh * 3;
  int ihc = imin(ho * 2, H - 1);
  int row = imin(imax(ihc + kh - 1, 0), H - 1);
  float phi = (((float)row + 0.5f) / (float)H) * 3.14159265358979f - 1.57079632679490f;
  float cs = fmaxf(fabsf(cosf(phi)), 1e-3f);
  float dw = fminf(0.8f / cs, 4.0f);
  int jw = (wo * 2) % W;
  float lf = roundf((float)jw + dw * (float)(kw - 1));
  int lo = (int)lf % W; if (lo < 0) lo += W;
  *lat = row; *lon = lo;
}

// x: f32 NCHW (256,6,64,128) -> xu: f16 NHWC (256,96,192,6)
__global__ void gather0_kernel(const float* __restrict__ x, _Float16* __restrict__ xu) {
  const int Hu = 96, Wu = 192, H = 64, W = 128, C = 6;
  int t = blockIdx.x * blockDim.x + threadIdx.x;
  if (t >= 256 * Hu * Wu) return;
  int j = t % Wu; int t2 = t / Wu; int i = t2 % Hu; int b = t2 / Hu;
  int lat, lon; sphere_idx(i, j, H, W, Wu, &lat, &lon);
  const float* src = x + ((size_t)(b * C) * H + lat) * W + lon;
  _Float16* dst = xu + ((size_t)(b * Hu + i) * Wu + j) * C;
#pragma unroll
  for (int c = 0; c < C; ++c) dst[c] = (_Float16)src[(size_t)c * H * W];
}

// act0: f16 NHWC (256,31,63,32) -> xu: f16 NHWC (256,48,96,32)
__global__ void gather1_kernel(const _Float16* __restrict__ a, _Float16* __restrict__ xu) {
  const int Hu = 48, Wu = 96, H = 31, W = 63, C = 32;
  int t = blockIdx.x * blockDim.x + threadIdx.x;
  if (t >= 256 * Hu * Wu) return;
  int j = t % Wu; int t2 = t / Wu; int i = t2 % Hu; int b = t2 / Hu;
  int lat, lon; sphere_idx(i, j, H, W, Wu, &lat, &lon);
  const _Float16* src = a + ((size_t)(b * H + lat) * W + lon) * C;
  _Float16* dst = xu + ((size_t)(b * Hu + i) * Wu + j) * C;
#pragma unroll
  for (int c = 0; c < C; ++c) dst[c] = src[c];
}

// ---------------------------------------------------------------------------
// WMMA fragment helpers (contiguous k = 16*h + s for A and B, wave32)
// ---------------------------------------------------------------------------
static __device__ __forceinline__ v16h load_a_frag(const _Float16* arow, int hsel) {
  v16h a;
#pragma unroll
  for (int p = 0; p < 8; ++p) {
    int klo = 2 * p + 16 * hsel;
    a[2 * p]     = arow[klo];
    a[2 * p + 1] = arow[klo + 1];
  }
  return a;
}
static __device__ __forceinline__ v16h load_b_frag(const _Float16* brow, int hsel) {
  v16h b;
#pragma unroll
  for (int e = 0; e < 16; ++e) b[e] = brow[e + 16 * hsel];
  return b;
}

// ---------------------------------------------------------------------------
// Implicit-GEMM conv, templated geometry. Block tile 128(M) x 32(N), 8 waves,
// each wave 16x32: two v_wmma_f32_16x16x32_f16 per K-step sharing one A frag.
// Ping-pong LDS double buffering: tile i+1's global loads are issued before
// tile i's WMMAs, so HBM latency is absorbed by the matrix math.
// Fused BN + LeakyReLU epilogue, f16 NHWC out.
// ---------------------------------------------------------------------------
template <int Hin, int Win, int Cin, int Hout, int Wout, int Cout,
          int KH, int KW, int S>
__global__ __launch_bounds__(256) void conv_wmma_kernel(
    const _Float16* __restrict__ in, const _Float16* __restrict__ w,
    const float* __restrict__ bnInv, const float* __restrict__ bnBeta,
    _Float16* __restrict__ out) {
  constexpr bool PACKED = (Cin % 32 == 0);
  constexpr int Ktot = PACKED ? (KH * KW * Cin) : (KH * 32);
  constexpr int NK = Ktot / 32;
  static_assert(PACKED || (KW * Cin <= 32), "pad-row path needs KW*Cin<=32");
  __shared__ _Float16 Atile[2][128][40];  // 80B row stride: 16B-aligned, bank-clean
  __shared__ _Float16 Btile[2][32][40];
  const int t = threadIdx.x;
  const int lane = t & 31, wv = t >> 5;
  const int m0 = blockIdx.x * 128;
  const int n0 = blockIdx.y * 32;
  const int nl = lane & 15, hsel = lane >> 4;

  // A: one row per thread pair, 16 halves per thread.
  const int arow_id = t >> 1;
  const int ahalf = (t & 1) * 16;
  long rowBase;
  {
    int mg = m0 + arow_id;              // grid exact: always < 256*Hout*Wout
    int wo = mg % Wout; int t2 = mg / Wout;
    int ho = t2 % Hout; int b = t2 / Hout;
    rowBase = ((long)(b * Hin + ho * S) * Win + wo * S) * Cin;
  }
  // B: 8 threads per row, 4 halves per thread.
  const int brow_id = t >> 3;
  const int bquad = (t & 7) * 4;

  uint4 aq0 = {}, aq1 = {};   // PACKED staging regs
  unsigned av[8];             // PADROW staging regs
  uint2 bq = {};

  auto load_tile = [&](int kk0) {
    if constexpr (PACKED) {
      int cell = kk0 / Cin, ci0 = kk0 % Cin;   // constant divides
      int kh = cell / KW, kw = cell - kh * KW;
      int off0 = (kh * Win + kw) * Cin + ci0 + ahalf;
      const uint4* src = (const uint4*)(in + rowBase + off0);
      aq0 = src[0];
      aq1 = src[1];
    } else {
      int kh = kk0 >> 5;
      const _Float16* src = in + rowBase + kh * (Win * Cin);
#pragma unroll
      for (int jj = 0; jj < 8; ++jj) {
        int idx = ahalf + 2 * jj;
        av[jj] = (idx < KW * Cin) ? *(const unsigned*)(src + idx) : 0u;
      }
    }
    bq = *(const uint2*)(w + (size_t)(n0 + brow_id) * Ktot + kk0 + bquad);
  };
  auto store_tile = [&](int buf) {
    if constexpr (PACKED) {
      *(uint4*)&Atile[buf][arow_id][ahalf]     = aq0;
      *(uint4*)&Atile[buf][arow_id][ahalf + 8] = aq1;
    } else {
#pragma unroll
      for (int jj = 0; jj < 8; ++jj)
        *(unsigned*)&Atile[buf][arow_id][ahalf + 2 * jj] = av[jj];
    }
    *(uint2*)&Btile[buf][brow_id][bquad] = bq;
  };

  v8f c0 = {0.f, 0.f, 0.f, 0.f, 0.f, 0.f, 0.f, 0.f};
  v8f c1 = {0.f, 0.f, 0.f, 0.f, 0.f, 0.f, 0.f, 0.f};

  load_tile(0);
  store_tile(0);
  __syncthreads();

#pragma unroll
  for (int i = 0; i < NK; ++i) {
    const int cur = i & 1;
    if (i + 1 < NK) load_tile((i + 1) * 32);   // in flight during compute
    v16h a  = load_a_frag(&Atile[cur][wv * 16 + nl][0], hsel);
    v16h b0 = load_b_frag(&Btile[cur][nl][0], hsel);
    v16h b1 = load_b_frag(&Btile[cur][16 + nl][0], hsel);
    c0 = __builtin_amdgcn_wmma_f32_16x16x32_f16(false, a, false, b0,
                                                (short)0, c0, false, false);
    c1 = __builtin_amdgcn_wmma_f32_16x16x32_f16(false, a, false, b1,
                                                (short)0, c1, false, false);
    if (i + 1 < NK) {
      __syncthreads();          // everyone done reading buffer 1-cur
      store_tile(1 - cur);      // loadcnt wait lands here, after the WMMAs
      __syncthreads();          // stores visible before next compute
    }
  }

  // Epilogue: BN + LeakyReLU, f16 NHWC store.
  const int ng0 = n0 + nl, ng1 = n0 + 16 + nl;
  float inv0 = bnInv[ng0],  bet0 = bnBeta[ng0];
  float inv1 = bnInv[ng1],  bet1 = bnBeta[ng1];
#pragma unroll
  for (int r = 0; r < 8; ++r) {
    int mg = m0 + wv * 16 + r + 8 * hsel;
    _Float16* orow = out + (size_t)mg * Cout;
    orow[ng0] = (_Float16)lrelu_f(c0[r] * inv0 + bet0);
    orow[ng1] = (_Float16)lrelu_f(c1[r] * inv1 + bet1);
  }
}

// ---------------------------------------------------------------------------
// Dense GEMM: out = A[M][K](f16) * Bw[N][K]^T (+bias)(+add)(lrelu) -> f32/f16
// Block tile 128x32, wave tile 16x32, ping-pong LDS pipeline.
// M%128==0, N%32==0, K%32==0, rows 16B-aligned.
// ---------------------------------------------------------------------------
__global__ __launch_bounds__(256) void gemm_wmma_kernel(
    const _Float16* __restrict__ A, const _Float16* __restrict__ Bw,
    const float* __restrict__ bias, const float* __restrict__ add,
    float* __restrict__ outF, _Float16* __restrict__ outH,
    int M, int N, int K, int doLrelu) {
  __shared__ _Float16 Atile[2][128][40];
  __shared__ _Float16 Btile[2][32][40];
  const int t = threadIdx.x;
  const int lane = t & 31, wv = t >> 5;
  const int m0 = blockIdx.x * 128;
  const int n0 = blockIdx.y * 32;
  const int nl = lane & 15, hsel = lane >> 4;
  const int arow_id = t >> 1;
  const int ahalf = (t & 1) * 16;
  const int brow_id = t >> 3;
  const int bquad = (t & 7) * 4;

  uint4 aq0 = {}, aq1 = {};
  uint2 bq = {};
  auto load_tile = [&](int kk0) {
    const uint4* src = (const uint4*)(A + (size_t)(m0 + arow_id) * K + kk0 + ahalf);
    aq0 = src[0];
    aq1 = src[1];
    bq = *(const uint2*)(Bw + (size_t)(n0 + brow_id) * K + kk0 + bquad);
  };
  auto store_tile = [&](int buf) {
    *(uint4*)&Atile[buf][arow_id][ahalf]     = aq0;
    *(uint4*)&Atile[buf][arow_id][ahalf + 8] = aq1;
    *(uint2*)&Btile[buf][brow_id][bquad] = bq;
  };

  v8f c0 = {0.f, 0.f, 0.f, 0.f, 0.f, 0.f, 0.f, 0.f};
  v8f c1 = {0.f, 0.f, 0.f, 0.f, 0.f, 0.f, 0.f, 0.f};

  load_tile(0);
  store_tile(0);
  __syncthreads();

  const int nk = K >> 5;
  for (int i = 0; i < nk; ++i) {
    const int cur = i & 1;
    if (i + 1 < nk) load_tile((i + 1) << 5);
    v16h a  = load_a_frag(&Atile[cur][wv * 16 + nl][0], hsel);
    v16h b0 = load_b_frag(&Btile[cur][nl][0], hsel);
    v16h b1 = load_b_frag(&Btile[cur][16 + nl][0], hsel);
    c0 = __builtin_amdgcn_wmma_f32_16x16x32_f16(false, a, false, b0,
                                                (short)0, c0, false, false);
    c1 = __builtin_amdgcn_wmma_f32_16x16x32_f16(false, a, false, b1,
                                                (short)0, c1, false, false);
    if (i + 1 < nk) {
      __syncthreads();
      store_tile(1 - cur);
      __syncthreads();
    }
  }

  const int ng0 = n0 + nl, ng1 = n0 + 16 + nl;
#pragma unroll
  for (int r = 0; r < 8; ++r) {
    int mg = m0 + wv * 16 + r + 8 * hsel;
    float a0 = c0[r], a1 = c1[r];
    if (bias) { a0 += bias[ng0]; a1 += bias[ng1]; }
    if (add)  { a0 += add[(size_t)mg * N + ng0]; a1 += add[(size_t)mg * N + ng1]; }
    if (doLrelu) { a0 = lrelu_f(a0); a1 = lrelu_f(a1); }
    if (outF) { outF[(size_t)mg * N + ng0] = a0; outF[(size_t)mg * N + ng1] = a1; }
    if (outH) { outH[(size_t)mg * N + ng0] = (_Float16)a0;
                outH[(size_t)mg * N + ng1] = (_Float16)a1; }
  }
}

// ---------------------------------------------------------------------------
// Small VALU kernels
// ---------------------------------------------------------------------------
__global__ void vfeat_kernel(const float* __restrict__ v, const float* __restrict__ vw,
                             const float* __restrict__ vb, float* __restrict__ vf) {
  int t = blockIdx.x * blockDim.x + threadIdx.x;
  if (t >= 65536) return;
  int n = t & 255, b = t >> 8;
  float s = vb[n];
#pragma unroll
  for (int k = 0; k < 9; ++k) s += v[b * 9 + k] * vw[n * 9 + k];
  vf[t] = s;
}

__global__ void gru_gate_kernel(const float* __restrict__ gi, const float* __restrict__ gh,
                                const float* __restrict__ hx, float* __restrict__ hxnew,
                                _Float16* __restrict__ lr16) {
  int t = blockIdx.x * blockDim.x + threadIdx.x;
  if (t >= 65536) return;
  int j = t & 255, b = t >> 8;
  const float* gib = gi + (size_t)b * 768;
  const float* ghb = gh + (size_t)b * 768;
  float r = 1.f / (1.f + expf(-(gib[j] + ghb[j])));
  float z = 1.f / (1.f + expf(-(gib[256 + j] + ghb[256 + j])));
  float n = tanhf(gib[512 + j] + r * ghb[512 + j]);
  float h = (1.f - z) * n + z * hx[t];
  hxnew[t] = h;
  lr16[t] = (_Float16)lrelu_f(h);
}

__global__ void fc_kernel(const _Float16* __restrict__ lr, const float* __restrict__ fw,
                          float* __restrict__ act) {
  int t = blockIdx.x * blockDim.x + threadIdx.x;
  if (t >= 1024) return;
  int o = t & 3, b = t >> 2;
  float s = 0.f;
  for (int k = 0; k < 256; ++k) s += (float)lr[b * 256 + k] * fw[o * 256 + k];
  act[t] = s;
}

// ---------------------------------------------------------------------------
extern "C" void kernel_launch(void* const* d_in, const int* in_sizes, int n_in,
                              void* d_out, int out_size, void* d_ws, size_t ws_size,
                              hipStream_t stream) {
  (void)in_sizes; (void)n_in; (void)out_size; (void)ws_size;
  const float* x      = (const float*)d_in[0];
  const float* vin    = (const float*)d_in[1];
  const float* hx     = (const float*)d_in[2];
  const float* sc0_w  = (const float*)d_in[3];
  const float* bn0p[4] = {(const float*)d_in[4], (const float*)d_in[5], (const float*)d_in[6], (const float*)d_in[7]};
  const float* sc1_w  = (const float*)d_in[8];
  const float* bn1p[4] = {(const float*)d_in[9], (const float*)d_in[10], (const float*)d_in[11], (const float*)d_in[12]};
  const float* c2_w   = (const float*)d_in[13];
  const float* bn2p[4] = {(const float*)d_in[14], (const float*)d_in[15], (const float*)d_in[16], (const float*)d_in[17]};
  const float* c3_w   = (const float*)d_in[18];
  const float* bn3p[4] = {(const float*)d_in[19], (const float*)d_in[20], (const float*)d_in[21], (const float*)d_in[22]};
  const float* c4_w   = (const float*)d_in[23];
  const float* bn4p[4] = {(const float*)d_in[24], (const float*)d_in[25], (const float*)d_in[26], (const float*)d_in[27]};
  const float* c5_w   = (const float*)d_in[28];
  const float* bn5p[4] = {(const float*)d_in[29], (const float*)d_in[30], (const float*)d_in[31], (const float*)d_in[32]};
  const float* proj_w = (const float*)d_in[33];
  const float* vproj_w = (const float*)d_in[34];
  const float* vproj_b = (const float*)d_in[35];
  const float* gru_wih = (const float*)d_in[36];
  const float* gru_whh = (const float*)d_in[37];
  const float* gru_bih = (const float*)d_in[38];
  const float* gru_bhh = (const float*)d_in[39];
  const float* fc_w    = (const float*)d_in[40];

  char* ws = (char*)d_ws;
  size_t off = 0;
  auto alloc = [&](size_t bytes) -> void* {
    void* p = ws + off;
    off += (bytes + 255) & ~(size_t)255;
    return p;
  };

  _Float16* W0    = (_Float16*)alloc(32  * 160  * 2);   // padded K=160
  _Float16* W1    = (_Float16*)alloc(64  * 800  * 2);
  _Float16* W2    = (_Float16*)alloc(64  * 576  * 2);
  _Float16* W3    = (_Float16*)alloc(64  * 256  * 2);
  _Float16* W4    = (_Float16*)alloc(128 * 576  * 2);
  _Float16* W5    = (_Float16*)alloc(128 * 1152 * 2);
  _Float16* PROJT = (_Float16*)alloc(256 * 2560 * 2);
  _Float16* WIH   = (_Float16*)alloc(768 * 256 * 2);
  _Float16* WHH   = (_Float16*)alloc(768 * 256 * 2);
  _Float16* HX16  = (_Float16*)alloc(256 * 256 * 2);
  float* INV[6]; float* BET[6];
  const int Cs[6] = {32, 64, 64, 64, 128, 128};
  for (int l = 0; l < 6; ++l) {
    INV[l] = (float*)alloc(Cs[l] * 4);
    BET[l] = (float*)alloc(Cs[l] * 4);
  }
  float*    VFEAT = (float*)alloc(65536 * 4);
  _Float16* FUSED = (_Float16*)alloc(65536 * 2);
  float*    GI    = (float*)alloc((size_t)256 * 768 * 4);
  float*    GH    = (float*)alloc((size_t)256 * 768 * 4);
  _Float16* LR16  = (_Float16*)alloc(65536 * 2);
  char* P0 = (char*)alloc((size_t)80 * 1024 * 1024);  // xu0/xu1/conv2/conv4
  char* P1 = (char*)alloc((size_t)36 * 1024 * 1024);  // conv0/conv1/conv3/conv5

  // ---- prep: BN folds + weight transforms ----
  const float** bns[6] = {bn0p, bn1p, bn2p, bn3p, bn4p, bn5p};
  for (int l = 0; l < 6; ++l)
    bn_prep_kernel<<<(Cs[l] + 63) / 64, 64, 0, stream>>>(bns[l][0], bns[l][1],
                                                         bns[l][2], bns[l][3],
                                                         INV[l], BET[l], Cs[l]);
  wt_sphere0_kernel<<<(32 * 5 * 32 + 255) / 256, 256, 0, stream>>>(sc0_w, W0);
  wt_sphere_kernel<<<(64 * 25 * 32 + 255) / 256, 256, 0, stream>>>(sc1_w, W1, 64, 32);
  wt_conv_kernel<<<(64 * 64 * 9 + 255) / 256, 256, 0, stream>>>(c2_w, W2, 64, 64, 3, 3);
  wt_conv_kernel<<<(64 * 64 * 4 + 255) / 256, 256, 0, stream>>>(c3_w, W3, 64, 64, 2, 2);
  wt_conv_kernel<<<(128 * 64 * 9 + 255) / 256, 256, 0, stream>>>(c4_w, W4, 128, 64, 3, 3);
  wt_conv_kernel<<<(128 * 128 * 9 + 255) / 256, 256, 0, stream>>>(c5_w, W5, 128, 128, 3, 3);
  wt_proj_kernel<<<(256 * 2560 + 255) / 256, 256, 0, stream>>>(proj_w, PROJT);
  f32_to_f16_kernel<<<(768 * 256 + 255) / 256, 256, 0, stream>>>(gru_wih, WIH, 768 * 256);
  f32_to_f16_kernel<<<(768 * 256 + 255) / 256, 256, 0, stream>>>(gru_whh, WHH, 768 * 256);
  f32_to_f16_kernel<<<(65536 + 255) / 256, 256, 0, stream>>>(hx, HX16, 65536);

  // ---- layer 0: gather + fused 5x5/stride3 conv + BN + lrelu ----
  gather0_kernel<<<(256 * 96 * 192 + 255) / 256, 256, 0, stream>>>(x, (_Float16*)P0);
  conv_wmma_kernel<96, 192, 6, 31, 63, 32, 5, 5, 3>
      <<<dim3(499968 / 128, 1), 256, 0, stream>>>(
          (const _Float16*)P0, W0, INV[0], BET[0], (_Float16*)P1);

  // ---- layer 1 ----
  gather1_kernel<<<(256 * 48 * 96 + 255) / 256, 256, 0, stream>>>((const _Float16*)P1,
                                                                  (_Float16*)P0);
  conv_wmma_kernel<48, 96, 32, 15, 31, 64, 5, 5, 3>
      <<<dim3(119040 / 128, 2), 256, 0, stream>>>(
          (const _Float16*)P0, W1, INV[1], BET[1], (_Float16*)P1);

  // ---- conv2..conv5 ----
  conv_wmma_kernel<15, 31, 64, 13, 29, 64, 3, 3, 1>
      <<<dim3(96512 / 128, 2), 256, 0, stream>>>(
          (const _Float16*)P1, W2, INV[2], BET[2], (_Float16*)P0);
  conv_wmma_kernel<13, 29, 64, 6, 14, 64, 2, 2, 2>
      <<<dim3(21504 / 128, 2), 256, 0, stream>>>(
          (const _Float16*)P0, W3, INV[3], BET[3], (_Float16*)P1);
  conv_wmma_kernel<6, 14, 64, 4, 12, 128, 3, 3, 1>
      <<<dim3(12288 / 128, 4), 256, 0, stream>>>(
          (const _Float16*)P1, W4, INV[4], BET[4], (_Float16*)P0);
  conv_wmma_kernel<4, 12, 128, 2, 10, 128, 3, 3, 1>
      <<<dim3(5120 / 128, 4), 256, 0, stream>>>(
          (const _Float16*)P0, W5, INV[5], BET[5], (_Float16*)P1);

  // ---- head: proj (+vproj fused) -> GRU -> fc ----
  vfeat_kernel<<<256, 256, 0, stream>>>(vin, vproj_w, vproj_b, VFEAT);
  gemm_wmma_kernel<<<dim3(2, 8), 256, 0, stream>>>(
      (const _Float16*)P1, PROJT, nullptr, VFEAT, nullptr, FUSED, 256, 256, 2560, 1);
  gemm_wmma_kernel<<<dim3(2, 24), 256, 0, stream>>>(
      FUSED, WIH, gru_bih, nullptr, GI, nullptr, 256, 768, 256, 0);
  gemm_wmma_kernel<<<dim3(2, 24), 256, 0, stream>>>(
      HX16, WHH, gru_bhh, nullptr, GH, nullptr, 256, 768, 256, 0);

  float* outAct = (float*)d_out;          // (256,4)
  float* outHx  = outAct + 1024;          // (256,256)
  gru_gate_kernel<<<256, 256, 0, stream>>>(GI, GH, hx, outHx, LR16);
  fc_kernel<<<4, 256, 0, stream>>>(LR16, fc_w, outAct);
}
